// MixerModel_57432302682330
// MI455X (gfx1250) — matrix operand order repaired and checked by reading, not compiled, not run on hardware
//
#include <hip/hip_runtime.h>

// ---------------- problem constants ----------------
#define BB   2
#define LL   1024
#define DM   768
#define DI   1536
#define DS   16
#define DC   4
#define DTR  48
#define XD   (DTR + 2 * DS)   // 80
#define ROWS (BB * LL)        // 2048
#define NL   4
#define EPSV 1e-5f

// LDS panel geometry for the staged GEMM
#define RS    40              // row stride in bf16 elements (80B, 16B aligned, breaks bank pattern)
#define PANEL (128 * RS)      // elements per panel buffer

typedef __attribute__((ext_vector_type(16))) __bf16 v16bf;
typedef __attribute__((ext_vector_type(8)))  float  v8f;

union Frag { unsigned int u[8]; v16bf v; };

static __device__ inline unsigned int bfbits(__bf16 x) {
    union { __bf16 b; unsigned short u; } c; c.b = x; return (unsigned int)c.u;
}

// ---------------- fp32 -> bf16 convert ----------------
__global__ __launch_bounds__(256)
void cvt_f32_bf16(const float* __restrict__ src, __bf16* __restrict__ dst, int n) {
    int i = blockIdx.x * blockDim.x + threadIdx.x;
    if (i < n) dst[i] = (__bf16)src[i];
}

// ---------------- fused residual-add + LayerNorm -> bf16 ----------------
__global__ __launch_bounds__(256)
void add_ln_bf16(const float* __restrict__ hidden, const float* __restrict__ residual,
                 int has_res, const float* __restrict__ w, const float* __restrict__ bias,
                 float* __restrict__ res_new, __bf16* __restrict__ nhb) {
    __shared__ float red[256];
    const int row = blockIdx.x;
    const float* hr = hidden   + (size_t)row * DM;
    const float* rr = residual + (size_t)row * DM;
    float vals[3];
    float s = 0.f;
    #pragma unroll
    for (int j = 0; j < 3; ++j) {
        int c = threadIdx.x + j * 256;
        float v = hr[c] + (has_res ? rr[c] : 0.f);
        vals[j] = v;
        res_new[(size_t)row * DM + c] = v;
        s += v;
    }
    red[threadIdx.x] = s; __syncthreads();
    for (int off = 128; off > 0; off >>= 1) {
        if (threadIdx.x < off) red[threadIdx.x] += red[threadIdx.x + off];
        __syncthreads();
    }
    const float mu = red[0] * (1.f / DM);
    __syncthreads();
    float s2 = 0.f;
    #pragma unroll
    for (int j = 0; j < 3; ++j) { float d = vals[j] - mu; s2 += d * d; }
    red[threadIdx.x] = s2; __syncthreads();
    for (int off = 128; off > 0; off >>= 1) {
        if (threadIdx.x < off) red[threadIdx.x] += red[threadIdx.x + off];
        __syncthreads();
    }
    const float rinv = rsqrtf(red[0] * (1.f / DM) + EPSV);
    #pragma unroll
    for (int j = 0; j < 3; ++j) {
        int c = threadIdx.x + j * 256;
        nhb[(size_t)row * DM + c] = (__bf16)((vals[j] - mu) * rinv * w[c] + bias[c]);
    }
}

// ---- async stage: copy 128x32 bf16 panels of A and W into LDS via ASYNC path ----
// 512 16B-chunks per panel; 256 threads x 2 chunks each; 4 async ops per thread total
static __device__ inline void stage_panel(const __bf16* gA, const __bf16* gW,
                                          unsigned ldsA, unsigned ldsW,
                                          int K, int k0, int tid) {
    #pragma unroll
    for (int j = 0; j < 2; ++j) {
        int c   = tid + j * 256;          // 0..511
        int row = c >> 2;                 // 0..127
        int kc  = (c & 3) << 3;           // 0,8,16,24 (elements)
        unsigned long long ga = (unsigned long long)(gA + (size_t)row * K + k0 + kc);
        unsigned la = ldsA + (unsigned)(row * (RS * 2) + kc * 2);
        asm volatile("global_load_async_to_lds_b128 %0, %1, off"
                     :: "v"(la), "v"(ga) : "memory");
    }
    #pragma unroll
    for (int j = 0; j < 2; ++j) {
        int c   = tid + j * 256;
        int row = c >> 2;
        int kc  = (c & 3) << 3;
        unsigned long long ga = (unsigned long long)(gW + (size_t)row * K + k0 + kc);
        unsigned la = ldsW + (unsigned)(row * (RS * 2) + kc * 2);
        asm volatile("global_load_async_to_lds_b128 %0, %1, off"
                     :: "v"(la), "v"(ga) : "memory");
    }
}

// ---------------- LDS-staged, double-buffered WMMA GEMM ----------------
// block = 8 waves = 128x128 tile; wave = 64x32 register tile (8 wmma / k-step)
// C[M,N] = A[M,K] * W[N,K]^T ; requires M%128==0, N%128==0, K%32==0
__global__ __launch_bounds__(256)
void gemm_bf16_wmma_lds(const __bf16* __restrict__ A, const __bf16* __restrict__ W,
                        float* __restrict__ C, int M, int N, int K) {
    __shared__ __bf16 Abuf[2 * PANEL];
    __shared__ __bf16 Wbuf[2 * PANEL];
    const int tid  = threadIdx.x;
    const int lane = tid & 31;
    const int wid  = tid >> 5;          // 0..7
    const int wm   = wid >> 2;          // 0..1  (64-row slice)
    const int wn   = wid & 3;           // 0..3  (32-col slice)
    const int nb   = N >> 7;
    const int bm0  = (blockIdx.x / nb) << 7;
    const int bn0  = (blockIdx.x % nb) << 7;
    const int kg   = lane >> 4;
    const int ln   = lane & 15;
    const __bf16* gA = A + (size_t)bm0 * K;
    const __bf16* gW = W + (size_t)bn0 * K;
    const unsigned ldsA0 = (unsigned)(unsigned long long)&Abuf[0];
    const unsigned ldsW0 = (unsigned)(unsigned long long)&Wbuf[0];

    v8f acc[4][2] = {};
    stage_panel(gA, gW, ldsA0, ldsW0, K, 0, tid);

    for (int k0 = 0; k0 < K; k0 += 32) {
        const int  buf  = (k0 >> 5) & 1;
        const bool more = (k0 + 32) < K;
        if (more) {
            stage_panel(gA, gW, ldsA0 + (unsigned)((buf ^ 1) * PANEL * 2),
                                ldsW0 + (unsigned)((buf ^ 1) * PANEL * 2), K, k0 + 32, tid);
            if (k0 + 64 < K) {   // near-prefetch the panel after next (global_prefetch_b8)
                __builtin_prefetch(gA + (size_t)(wid * 16) * K + k0 + 64, 0, 3);
                __builtin_prefetch(gW + (size_t)(wid * 16) * K + k0 + 64, 0, 3);
            }
            asm volatile("s_wait_asynccnt 4" ::: "memory"); // own prev panel retired
        } else {
            asm volatile("s_wait_asynccnt 0" ::: "memory");
        }
        __syncthreads();        // all waves' staged data visible

        const __bf16* a0 = &Abuf[buf * PANEL];
        const __bf16* w0 = &Wbuf[buf * PANEL];
        Frag a[4], b[2];
        #pragma unroll
        for (int v = 0; v < 8; ++v) {
            // A 16x32 bf16 layout: VGPR v<4 -> K = kg*8 + 2v ; v>=4 -> +16
            int ka = ((v >> 2) << 4) + (kg << 3) + ((v & 3) << 1);
            // B 32x16 bf16 layout: same as A: VGPR v<4 -> K = kg*8 + 2v ; v>=4 -> +16
            int kb = ((v >> 2) << 4) + (kg << 3) + ((v & 3) << 1);
            #pragma unroll
            for (int mi = 0; mi < 4; ++mi)
                a[mi].u[v] = *(const unsigned int*)(a0 + (wm * 64 + mi * 16 + ln) * RS + ka);
            #pragma unroll
            for (int ni = 0; ni < 2; ++ni)
                b[ni].u[v] = *(const unsigned int*)(w0 + (wn * 32 + ni * 16 + ln) * RS + kb);
        }
        #pragma unroll
        for (int mi = 0; mi < 4; ++mi)
            #pragma unroll
            for (int ni = 0; ni < 2; ++ni)
                acc[mi][ni] = __builtin_amdgcn_wmma_f32_16x16x32_bf16(
                    false, a[mi].v, false, b[ni].v, (short)0, acc[mi][ni], false, false);
        __syncthreads();        // reads done before this buffer is restaged
    }

    #pragma unroll
    for (int mi = 0; mi < 4; ++mi) {
        #pragma unroll
        for (int ni = 0; ni < 2; ++ni) {
            float* crow = C + (size_t)(bm0 + wm * 64 + mi * 16 + (kg << 3)) * N
                            + bn0 + wn * 32 + ni * 16 + ln;
            #pragma unroll
            for (int v = 0; v < 8; ++v) crow[(size_t)v * N] = acc[mi][ni][v];
        }
    }
}

// ---------------- simple WMMA GEMM (one 16x16 tile / wave) for odd shapes ----------
template<bool KCHK>
__global__ __launch_bounds__(256)
void gemm_bf16_wmma(const __bf16* __restrict__ A, const __bf16* __restrict__ W,
                    float* __restrict__ C, int M, int N, int K) {
    const int lane = threadIdx.x & 31;
    const int tile = blockIdx.x * (blockDim.x >> 5) + (threadIdx.x >> 5);
    const int tn_cnt = N >> 4;
    const int tm = tile / tn_cnt;
    const int tn = tile - tm * tn_cnt;
    if (tm * 16 >= M) return;   // wave-uniform
    const int row = tm * 16 + (lane & 15);
    const int col = tn * 16 + (lane & 15);
    const int kg  = lane >> 4;
    const __bf16* arow = A + (size_t)row * K;
    const __bf16* wrow = W + (size_t)col * K;
    v8f acc = {};
    for (int k0 = 0; k0 < K; k0 += 32) {
        Frag a, b;
        #pragma unroll
        for (int v = 0; v < 8; ++v) {
            int ka = k0 + ((v >> 2) << 4) + (kg << 3) + ((v & 3) << 1);
            int kb = k0 + ((v >> 2) << 4) + (kg << 3) + ((v & 3) << 1);
            if (!KCHK) {
                a.u[v] = *(const unsigned int*)(arow + ka);
                b.u[v] = *(const unsigned int*)(wrow + kb);
            } else {
                unsigned int a0 = (ka     < K) ? bfbits(arow[ka])     : 0u;
                unsigned int a1 = (ka + 1 < K) ? bfbits(arow[ka + 1]) : 0u;
                unsigned int b0 = (kb     < K) ? bfbits(wrow[kb])     : 0u;
                unsigned int b1 = (kb + 1 < K) ? bfbits(wrow[kb + 1]) : 0u;
                a.u[v] = a0 | (a1 << 16);
                b.u[v] = b0 | (b1 << 16);
            }
        }
        acc = __builtin_amdgcn_wmma_f32_16x16x32_bf16(false, a.v, false, b.v,
                                                      (short)0, acc, false, false);
    }
    float* crow = C + (size_t)(tm * 16 + (kg << 3)) * N + col;
    #pragma unroll
    for (int v = 0; v < 8; ++v) crow[(size_t)v * N] = acc[v];
}

// ---------------- causal depthwise conv (D_CONV=4) + SiLU ----------------
__global__ __launch_bounds__(256)
void conv_silu(const float* __restrict__ xz, const float* __restrict__ cw,
               const float* __restrict__ cb, float* __restrict__ xc,
               __bf16* __restrict__ xcb) {
    int i = blockIdx.x * blockDim.x + threadIdx.x;
    if (i >= ROWS * DI) return;
    int d = i % DI;
    int r = i / DI;
    int l = r % LL;
    float acc = cb[d];
    #pragma unroll
    for (int j = 0; j < DC; ++j) {
        int ls = l - (DC - 1) + j;
        if (ls >= 0)
            acc += cw[d * DC + j] * xz[(size_t)(r - (DC - 1) + j) * (2 * DI) + d];
    }
    float s = acc / (1.f + __expf(-acc));
    xc[i]  = s;
    xcb[i] = (__bf16)s;
}

// ---------------- slice xdbl[:, :48] to bf16 ----------------
__global__ __launch_bounds__(256)
void slice_dt_bf16(const float* __restrict__ xdbl, __bf16* __restrict__ dst) {
    int i = blockIdx.x * blockDim.x + threadIdx.x;
    if (i >= ROWS * DTR) return;
    int r = i / DTR, c = i - r * DTR;
    dst[i] = (__bf16)xdbl[(size_t)r * XD + c];
}

// ---------------- dt = softplus(dt_pre + dt_bias), in-place ----------------
__global__ __launch_bounds__(256)
void dt_softplus(float* __restrict__ dt, const float* __restrict__ dtb) {
    int i = blockIdx.x * blockDim.x + threadIdx.x;
    if (i >= ROWS * DI) return;
    int d = i % DI;
    float v = dt[i] + dtb[d];
    dt[i] = (v > 20.f) ? v : log1pf(__expf(v));
}

// ---------------- selective scan + C-contraction + D-skip + z-gate ----------------
__global__ __launch_bounds__(256)
void scan_gate(const float* __restrict__ dt, const float* __restrict__ xc,
               const float* __restrict__ xdbl, const float* __restrict__ xz,
               const float* __restrict__ alog, const float* __restrict__ dp,
               __bf16* __restrict__ ybf) {
    int t = blockIdx.x * blockDim.x + threadIdx.x;
    if (t >= BB * DI) return;
    int b = t / DI, d = t - b * DI;
    float A[DS], h[DS];
    #pragma unroll
    for (int n = 0; n < DS; ++n) { A[n] = -__expf(alog[d * DS + n]); h[n] = 0.f; }
    const float dpd = dp[d];
    for (int l = 0; l < LL; ++l) {
        int r = b * LL + l;
        float dtv = dt[(size_t)r * DI + d];
        float xcv = xc[(size_t)r * DI + d];
        const float* bm = xdbl + (size_t)r * XD + DTR;
        const float* cm = bm + DS;
        float y = 0.f;
        #pragma unroll
        for (int n = 0; n < DS; ++n) {
            h[n] = __expf(dtv * A[n]) * h[n] + dtv * bm[n] * xcv;
            y = fmaf(h[n], cm[n], y);
        }
        y = fmaf(dpd, xcv, y);
        float z = xz[(size_t)r * (2 * DI) + DI + d];
        y *= z / (1.f + __expf(-z));
        ybf[(size_t)r * DI + d] = (__bf16)y;
    }
}

// ---------------- masked hidden/residual update ----------------
__global__ __launch_bounds__(256)
void merge_mask(const float* __restrict__ nh_out, const unsigned char* __restrict__ mask,
                const float* __restrict__ res_new, float* __restrict__ hidden,
                float* __restrict__ residual, int first_layer) {
    int i = blockIdx.x * blockDim.x + threadIdx.x;
    if (i >= ROWS * DM) return;
    int row = i / DM;
    bool m = mask[row] != 0;
    if (m) hidden[i] = nh_out[i];
    if (first_layer || m) residual[i] = res_new[i];
}

// ---------------- final add + LayerNorm -> f32 output ----------------
__global__ __launch_bounds__(256)
void final_ln(const float* __restrict__ hidden, const float* __restrict__ residual,
              const float* __restrict__ w, const float* __restrict__ bias,
              float* __restrict__ out) {
    __shared__ float red[256];
    const int row = blockIdx.x;
    const float* hr = hidden   + (size_t)row * DM;
    const float* rr = residual + (size_t)row * DM;
    float vals[3];
    float s = 0.f;
    #pragma unroll
    for (int j = 0; j < 3; ++j) {
        int c = threadIdx.x + j * 256;
        float v = hr[c] + rr[c];
        vals[j] = v;
        s += v;
    }
    red[threadIdx.x] = s; __syncthreads();
    for (int off = 128; off > 0; off >>= 1) {
        if (threadIdx.x < off) red[threadIdx.x] += red[threadIdx.x + off];
        __syncthreads();
    }
    const float mu = red[0] * (1.f / DM);
    __syncthreads();
    float s2 = 0.f;
    #pragma unroll
    for (int j = 0; j < 3; ++j) { float dlt = vals[j] - mu; s2 += dlt * dlt; }
    red[threadIdx.x] = s2; __syncthreads();
    for (int off = 128; off > 0; off >>= 1) {
        if (threadIdx.x < off) red[threadIdx.x] += red[threadIdx.x + off];
        __syncthreads();
    }
    const float rinv = rsqrtf(red[0] * (1.f / DM) + EPSV);
    #pragma unroll
    for (int j = 0; j < 3; ++j) {
        int c = threadIdx.x + j * 256;
        out[(size_t)row * DM + c] = (vals[j] - mu) * rinv * w[c] + bias[c];
    }
}

// =====================================================================
extern "C" void kernel_launch(void* const* d_in, const int* in_sizes, int n_in,
                              void* d_out, int out_size, void* d_ws, size_t ws_size,
                              hipStream_t stream) {
    (void)in_sizes; (void)n_in; (void)out_size; (void)ws_size;

    const float*         x_in      = (const float*)d_in[0];
    const unsigned char* masks     = (const unsigned char*)d_in[1];
    const float*         norm_w    = (const float*)d_in[2];
    const float*         norm_b    = (const float*)d_in[3];
    const float*         in_projw  = (const float*)d_in[4];
    const float*         conv_w    = (const float*)d_in[5];
    const float*         conv_b    = (const float*)d_in[6];
    const float*         x_projw   = (const float*)d_in[7];
    const float*         dt_projw  = (const float*)d_in[8];
    const float*         dt_projb  = (const float*)d_in[9];
    const float*         A_log     = (const float*)d_in[10];
    const float*         D_param   = (const float*)d_in[11];
    const float*         out_projw = (const float*)d_in[12];
    const float*         normf_w   = (const float*)d_in[13];
    const float*         normf_b   = (const float*)d_in[14];

    char* p = (char*)d_ws;
    auto alloc = [&](size_t bytes) -> void* {
        void* r = (void*)p;
        p += (bytes + 255) & ~(size_t)255;
        return r;
    };
    float*  hidden   = (float*)alloc(sizeof(float) * ROWS * DM);
    float*  residual = (float*)alloc(sizeof(float) * ROWS * DM);
    float*  res_new  = (float*)alloc(sizeof(float) * ROWS * DM);
    float*  xz       = (float*)alloc(sizeof(float) * ROWS * 2 * DI);
    float*  xc       = (float*)alloc(sizeof(float) * ROWS * DI);
    float*  xdbl     = (float*)alloc(sizeof(float) * ROWS * XD);
    float*  dtbuf    = (float*)alloc(sizeof(float) * ROWS * DI);
    float*  nh_out   = (float*)alloc(sizeof(float) * ROWS * DM);
    __bf16* nh_bf    = (__bf16*)alloc(sizeof(__bf16) * ROWS * DM);
    __bf16* xc_bf    = (__bf16*)alloc(sizeof(__bf16) * ROWS * DI);
    __bf16* dt48_bf  = (__bf16*)alloc(sizeof(__bf16) * ROWS * DTR);
    __bf16* y_bf     = (__bf16*)alloc(sizeof(__bf16) * ROWS * DI);
    __bf16* w_in_bf  = (__bf16*)alloc(sizeof(__bf16) * 2 * DI * DM);
    __bf16* w_xp_bf  = (__bf16*)alloc(sizeof(__bf16) * XD * DI);
    __bf16* w_dt_bf  = (__bf16*)alloc(sizeof(__bf16) * DI * DTR);
    __bf16* w_op_bf  = (__bf16*)alloc(sizeof(__bf16) * DM * DI);

    hipMemcpyAsync(hidden, x_in, sizeof(float) * ROWS * DM,
                   hipMemcpyDeviceToDevice, stream);

    const int TPB = 256;
    auto gemm_blocks_lds = [](int M, int N) { return (M / 128) * (N / 128); };
    auto gemm_blocks     = [](int M, int N) { return ((M / 16) * (N / 16) + 7) / 8; };

    for (int i = 0; i < NL; ++i) {
        {
            int n1 = 2 * DI * DM, n2 = XD * DI, n3 = DI * DTR, n4 = DM * DI;
            cvt_f32_bf16<<<(n1 + TPB - 1) / TPB, TPB, 0, stream>>>(in_projw  + (size_t)i * n1, w_in_bf, n1);
            cvt_f32_bf16<<<(n2 + TPB - 1) / TPB, TPB, 0, stream>>>(x_projw   + (size_t)i * n2, w_xp_bf, n2);
            cvt_f32_bf16<<<(n3 + TPB - 1) / TPB, TPB, 0, stream>>>(dt_projw  + (size_t)i * n3, w_dt_bf, n3);
            cvt_f32_bf16<<<(n4 + TPB - 1) / TPB, TPB, 0, stream>>>(out_projw + (size_t)i * n4, w_op_bf, n4);
        }
        add_ln_bf16<<<ROWS, TPB, 0, stream>>>(hidden, (i ? residual : hidden), i > 0,
                                              norm_w + (size_t)i * DM, norm_b + (size_t)i * DM,
                                              res_new, nh_bf);
        // xz = nh @ in_proj^T (2048 x 768 x 3072) -- async-LDS staged WMMA
        gemm_bf16_wmma_lds<<<gemm_blocks_lds(ROWS, 2 * DI), TPB, 0, stream>>>(
            nh_bf, w_in_bf, xz, ROWS, 2 * DI, DM);
        conv_silu<<<(ROWS * DI + TPB - 1) / TPB, TPB, 0, stream>>>(
            xz, conv_w + (size_t)i * DI * DC, conv_b + (size_t)i * DI, xc, xc_bf);
        // xdbl = xc @ x_proj^T (2048 x 1536 x 80) -- N=80 not 128-aligned
        gemm_bf16_wmma<false><<<gemm_blocks(ROWS, XD), TPB, 0, stream>>>(
            xc_bf, w_xp_bf, xdbl, ROWS, XD, DI);
        slice_dt_bf16<<<(ROWS * DTR + TPB - 1) / TPB, TPB, 0, stream>>>(xdbl, dt48_bf);
        // dt_pre = xdbl[:, :48] @ dt_proj^T (K=48 needs bounds checks)
        gemm_bf16_wmma<true><<<gemm_blocks(ROWS, DI), TPB, 0, stream>>>(
            dt48_bf, w_dt_bf, dtbuf, ROWS, DI, DTR);
        dt_softplus<<<(ROWS * DI + TPB - 1) / TPB, TPB, 0, stream>>>(
            dtbuf, dt_projb + (size_t)i * DI);
        scan_gate<<<(BB * DI + TPB - 1) / TPB, TPB, 0, stream>>>(
            dtbuf, xc, xdbl, xz, A_log + (size_t)i * DI * DS, D_param + (size_t)i * DI, y_bf);
        // nh_out = y @ out_proj^T (2048 x 1536 x 768) -- async-LDS staged WMMA
        gemm_bf16_wmma_lds<<<gemm_blocks_lds(ROWS, DM), TPB, 0, stream>>>(
            y_bf, w_op_bf, nh_out, ROWS, DM, DI);
        merge_mask<<<(ROWS * DM + TPB - 1) / TPB, TPB, 0, stream>>>(
            nh_out, masks + (size_t)i * ROWS, res_new, hidden, residual, i == 0);
    }

    final_ln<<<ROWS, TPB, 0, stream>>>(hidden, residual, normf_w, normf_b, (float*)d_out);
}
